// RHN_13838384628114
// MI455X (gfx1250) — matrix operand order repaired and checked by reading, not compiled, not run on hardware
//
#include <hip/hip_runtime.h>
#include <stdint.h>

typedef __attribute__((ext_vector_type(16))) _Float16 v16h;
typedef __attribute__((ext_vector_type(8)))  _Float16 v8h;
typedef __attribute__((ext_vector_type(8)))  float    v8f;

#define DD 512
#define TT 2048
#define LDS_STRIDE 520   // 512 + 8 halves pad -> 16B-aligned, bank-conflict-free fragment reads
#define MAT_ELEMS (512 * 512)
#define ROWS_PER_BLOCK 32

// ---------------- threefry2x32 (JAX-compatible) ----------------
__device__ __forceinline__ uint32_t rotl32(uint32_t x, int r) {
    return (x << r) | (x >> (32 - r));
}

__device__ __forceinline__ void threefry2x32(uint32_t k0, uint32_t k1,
                                             uint32_t c0, uint32_t c1,
                                             uint32_t& o0, uint32_t& o1) {
    const int rot[8] = {13, 15, 26, 6, 17, 29, 16, 24};
    uint32_t ks0 = k0, ks1 = k1, ks2 = k0 ^ k1 ^ 0x1BD11BDAu;
    uint32_t x0 = c0 + ks0, x1 = c1 + ks1;
#pragma unroll
    for (int g = 0; g < 5; ++g) {
#pragma unroll
        for (int i = 0; i < 4; ++i) {
            x0 += x1;
            x1 = rotl32(x1, rot[(g % 2) * 4 + i]);
            x1 ^= x0;
        }
        switch (g) {
            case 0: x0 += ks1; x1 += ks2 + 1u; break;
            case 1: x0 += ks2; x1 += ks0 + 2u; break;
            case 2: x0 += ks0; x1 += ks1 + 3u; break;
            case 3: x0 += ks1; x1 += ks2 + 4u; break;
            case 4: x0 += ks2; x1 += ks0 + 5u; break;
        }
    }
    o0 = x0; o1 = x1;
}

// noise_i: keep=0.25 over [32,512]; noise_s: keep=0.75 over [32,512]
__global__ void gen_noise_kernel(float* __restrict__ noise_i, float* __restrict__ noise_s) {
    unsigned idx = blockIdx.x * 256u + threadIdx.x;   // 0..32767
    if (idx >= 32768u) return;
    // split(key(42)) -> k1 (input noise), k2 (state noise)
    uint32_t a0, b0, a1, b1;
    threefry2x32(0u, 42u, 0u, 2u, a0, b0);
    threefry2x32(0u, 42u, 1u, 3u, a1, b1);
    bool is_s = idx >= 16384u;
    unsigned e = idx & 16383u;
    uint32_t k0 = is_s ? b0 : a0;
    uint32_t k1 = is_s ? b1 : a1;
    uint32_t o0, o1, bits;
    if (e < 8192u) { threefry2x32(k0, k1, e, e + 8192u, o0, o1); bits = o0; }
    else           { threefry2x32(k0, k1, e - 8192u, e, o0, o1); bits = o1; }
    float u = __uint_as_float((bits >> 9) | 0x3F800000u) - 1.0f;
    float keep = is_s ? 0.75f : 0.25f;
    float m = (u < keep) ? (1.0f / keep) : 0.0f;
    (is_s ? noise_s : noise_i)[e] = m;
}

// transpose + f32->f16: wt[mat][n*512 + k] = w[mat][k*512 + n]
__global__ void wconv_kernel(const float* __restrict__ w, _Float16* __restrict__ wt, int total) {
    int idx = blockIdx.x * 256 + threadIdx.x;
    if (idx >= total) return;
    int mat = idx >> 18;
    int r   = idx & (MAT_ELEMS - 1);
    int n   = r >> 9;
    int k   = r & 511;
    wt[idx] = (_Float16)w[(size_t)mat * MAT_ELEMS + (size_t)k * 512 + n];
}

// Layer-0 recurrent contribution: s0 = h0 is row-constant, so
// vec{H,T}[b][col] = sum_k h0[k]*noise_s[b][k]*rW0[k][col] + rb0[col]  (per batch, fp32)
__global__ void vec0_kernel(const float* __restrict__ h0, const float* __restrict__ noise_s,
                            const float* __restrict__ rHw0, const float* __restrict__ rHb0,
                            const float* __restrict__ rTw0, const float* __restrict__ rTb0,
                            float* __restrict__ vecH, float* __restrict__ vecT) {
    int b = blockIdx.x;                 // 0..31
    for (int c = threadIdx.x; c < DD; c += 256) {
        float ah = rHb0[c], at = rTb0[c];
        for (int k = 0; k < DD; ++k) {
            float sd = h0[k] * noise_s[b * DD + k];
            ah += sd * rHw0[(size_t)k * DD + c];
            at += sd * rTw0[(size_t)k * DD + c];
        }
        vecH[b * DD + c] = ah;
        vecT[b * DD + c] = at;
    }
}

// ---------------- fused RHN kernel ----------------
// GEMM sweep: acc[rt][j] += A(rt-th 16x512 tile in LDS) x Wt(cols nwave+j*16..+15), K=512.
// B fragments are loaded once and reused across both row-tiles (2x L2 traffic reduction).
__device__ __forceinline__ void gemm_sweep2(const _Float16* __restrict__ aLds,
                                            const _Float16* __restrict__ wt,
                                            int lane, int nwave, v8f acc[2][4]) {
    const int m    = lane & 15;
    const int half = lane >> 4;
    for (int k0 = 0; k0 < 512; k0 += 32) {
        v16h a[2];
#pragma unroll
        for (int rt = 0; rt < 2; ++rt) {
            // A fragment (16x32 f16, §7.12.2): lane holds K = half*8..+7 and 16+half*8..+7 of row m
            const _Float16* pa = aLds + (rt * 16 + m) * LDS_STRIDE + k0 + half * 8;
            v8h alo = *(const v8h*)pa;
            v8h ahi = *(const v8h*)(pa + 16);
#pragma unroll
            for (int i = 0; i < 8; ++i) { a[rt][i] = alo[i]; a[rt][8 + i] = ahi[i]; }
        }
#pragma unroll
        for (int j = 0; j < 4; ++j) {
            // B fragment (32x16): lane = col n, holds K = k0 + half*8..+7 and k0 + 16+half*8..+7 (in wt)
            const _Float16* pb = wt + (size_t)(nwave + j * 16 + m) * DD + k0 + half * 8;
            v8h blo = *(const v8h*)pb;
            v8h bhi = *(const v8h*)(pb + 16);
            v16h b;
#pragma unroll
            for (int i = 0; i < 8; ++i) { b[i] = blo[i]; b[8 + i] = bhi[i]; }
#pragma unroll
            for (int rt = 0; rt < 2; ++rt) {
                acc[rt][j] = __builtin_amdgcn_wmma_f32_16x16x32_f16(
                    false, a[rt], false, b, (short)0, acc[rt][j], false, false);
            }
        }
    }
}

__global__ __launch_bounds__(256, 1)
void rhn_fused_kernel(const float* __restrict__ seq, const float* __restrict__ h0,
                      const float* __restrict__ bH,  const float* __restrict__ bT,
                      const float* __restrict__ rHb, const float* __restrict__ rTb,
                      const _Float16* __restrict__ wtH,  const _Float16* __restrict__ wtT,
                      const _Float16* __restrict__ wtRH, const _Float16* __restrict__ wtRT,
                      const float* __restrict__ noise_i, const float* __restrict__ noise_s,
                      const float* __restrict__ vecH, const float* __restrict__ vecT,
                      float* __restrict__ out) {
    __shared__ __align__(16) _Float16 Atile[ROWS_PER_BLOCK * LDS_STRIDE];

    const int r0   = blockIdx.x * ROWS_PER_BLOCK;   // all rows in a block share batch b
    const int bIdx = r0 / TT;
    const int tid  = threadIdx.x;
    const int wv   = tid >> 5;
    const int lane = tid & 31;
    const int nwave = wv * 64;                      // this wave's 64 output columns
    const int half = lane >> 4;

    // Stage x-tile: seq * noise_i, f32 -> f16 into LDS (coalesced)
    const float* nIb = noise_i + bIdx * DD;
#pragma unroll
    for (int it = 0; it < (ROWS_PER_BLOCK * DD) / 256; ++it) {
        int e = tid + it * 256;
        int row = e >> 9, col = e & 511;
        float v = seq[(size_t)(r0 + row) * DD + col] * nIb[col];
        Atile[row * LDS_STRIDE + col] = (_Float16)v;
    }
    __syncthreads();

    // Per-lane column indices + per-column constants
    int colj[4]; float nsj[4];
    const float* nSb = noise_s + bIdx * DD;
#pragma unroll
    for (int j = 0; j < 4; ++j) {
        colj[j] = nwave + j * 16 + (lane & 15);
        nsj[j]  = nSb[colj[j]];
    }

    // ---- Layer 0: h_lin = x@wH + (bH + vecH[b]); t_lin = x@wT + (bT + vecT[b]) ----
    v8f hA[2][4], tA[2][4];
#pragma unroll
    for (int j = 0; j < 4; ++j) {
        float vh = bH[colj[j]] + vecH[bIdx * DD + colj[j]];
        float vt = bT[colj[j]] + vecT[bIdx * DD + colj[j]];
#pragma unroll
        for (int rt = 0; rt < 2; ++rt)
#pragma unroll
            for (int i = 0; i < 8; ++i) { hA[rt][j][i] = vh; tA[rt][j][i] = vt; }
    }
    gemm_sweep2(Atile, wtH, lane, nwave, hA);
    gemm_sweep2(Atile, wtT, lane, nwave, tA);

    v8f s[2][4];
#pragma unroll
    for (int j = 0; j < 4; ++j) {
        float h0v = h0[colj[j]];
#pragma unroll
        for (int rt = 0; rt < 2; ++rt)
#pragma unroll
            for (int i = 0; i < 8; ++i) {
                float Hg = tanhf(hA[rt][j][i]);
                float Tg = 1.0f / (1.0f + __expf(-tA[rt][j][i]));
                s[rt][j][i] = (Hg - h0v) * Tg + h0v;
            }
    }

    // ---- Layers 1..2: full recurrent GEMMs over sd = s * noise_s ----
    for (int l = 1; l < 3; ++l) {
        __syncthreads();   // previous A-tile fully consumed by all waves
#pragma unroll
        for (int rt = 0; rt < 2; ++rt)
#pragma unroll
            for (int j = 0; j < 4; ++j)
#pragma unroll
                for (int i = 0; i < 8; ++i) {
                    int m = rt * 16 + i + half * 8;          // C-layout row
                    Atile[m * LDS_STRIDE + colj[j]] = (_Float16)(s[rt][j][i] * nsj[j]);
                }
        __syncthreads();

        const float* rhb = rHb + l * DD;
        const float* rtb = rTb + l * DD;
#pragma unroll
        for (int j = 0; j < 4; ++j) {
            float vh = rhb[colj[j]], vt = rtb[colj[j]];
#pragma unroll
            for (int rt = 0; rt < 2; ++rt)
#pragma unroll
                for (int i = 0; i < 8; ++i) { hA[rt][j][i] = vh; tA[rt][j][i] = vt; }
        }
        gemm_sweep2(Atile, wtRH + (size_t)(l - 1) * MAT_ELEMS, lane, nwave, hA);
        gemm_sweep2(Atile, wtRT + (size_t)(l - 1) * MAT_ELEMS, lane, nwave, tA);

#pragma unroll
        for (int rt = 0; rt < 2; ++rt)
#pragma unroll
            for (int j = 0; j < 4; ++j)
#pragma unroll
                for (int i = 0; i < 8; ++i) {
                    float Hg = tanhf(hA[rt][j][i]);
                    float Tg = 1.0f / (1.0f + __expf(-tA[rt][j][i]));
                    s[rt][j][i] = (Hg - s[rt][j][i]) * Tg + s[rt][j][i];
                }
    }

    // Store s (f32) to d_out
#pragma unroll
    for (int rt = 0; rt < 2; ++rt)
#pragma unroll
        for (int j = 0; j < 4; ++j)
#pragma unroll
            for (int i = 0; i < 8; ++i) {
                int m = rt * 16 + i + half * 8;
                out[(size_t)(r0 + m) * DD + colj[j]] = s[rt][j][i];
            }
}

extern "C" void kernel_launch(void* const* d_in, const int* in_sizes, int n_in,
                              void* d_out, int out_size, void* d_ws, size_t ws_size,
                              hipStream_t stream) {
    const float* h0  = (const float*)d_in[0];
    const float* seq = (const float*)d_in[1];
    const float* wH  = (const float*)d_in[2];
    const float* bH  = (const float*)d_in[3];
    const float* wT  = (const float*)d_in[4];
    const float* bT  = (const float*)d_in[5];
    const float* rHw = (const float*)d_in[6];
    const float* rHb = (const float*)d_in[7];
    const float* rTw = (const float*)d_in[8];
    const float* rTb = (const float*)d_in[9];
    float* out = (float*)d_out;

    // Workspace: 6 f16 512x512 matrices (3 MB) + noise masks + layer-0 vectors
    _Float16* wtH  = (_Float16*)d_ws;
    _Float16* wtT  = wtH  + MAT_ELEMS;
    _Float16* wtRH = wtT  + MAT_ELEMS;        // recur layers 1..2
    _Float16* wtRT = wtRH + 2 * MAT_ELEMS;
    float* noise_i = (float*)(wtRT + 2 * MAT_ELEMS);
    float* noise_s = noise_i + 32 * DD;
    float* vecH    = noise_s + 32 * DD;
    float* vecT    = vecH    + 32 * DD;

    wconv_kernel<<<MAT_ELEMS / 256, 256, 0, stream>>>(wH, wtH, MAT_ELEMS);
    wconv_kernel<<<MAT_ELEMS / 256, 256, 0, stream>>>(wT, wtT, MAT_ELEMS);
    wconv_kernel<<<2 * MAT_ELEMS / 256, 256, 0, stream>>>(rHw + MAT_ELEMS, wtRH, 2 * MAT_ELEMS);
    wconv_kernel<<<2 * MAT_ELEMS / 256, 256, 0, stream>>>(rTw + MAT_ELEMS, wtRT, 2 * MAT_ELEMS);
    gen_noise_kernel<<<128, 256, 0, stream>>>(noise_i, noise_s);
    vec0_kernel<<<32, 256, 0, stream>>>(h0, noise_s, rHw, rHb, rTw, rTb, vecH, vecT);

    // 65536 rows / 32 rows-per-block = 2048 blocks of 8 wave32s
    rhn_fused_kernel<<<2048, 256, 0, stream>>>(seq, h0, bH, bT, rHb, rTb,
                                               wtH, wtT, wtRH, wtRT,
                                               noise_i, noise_s, vecH, vecT, out);
}
